// HCMamba_CSI_HPE_515396076441
// MI455X (gfx1250) — compile-verified, matrix operand order checked by reading, not run on hardware
//
#include <hip/hip_runtime.h>
#include <hip/hip_bf16.h>
#include <math.h>

// ---------------- model constants ----------------
#define BATCH   4
#define SEQ_L   2048
#define SEQ_L2  512          // after RATE=4 downsample
#define DM      256          // D_MODEL
#define DI      512          // D_INNER
#define DSN     16           // D_STATE
#define DTR     16           // DT_RANK
#define DCONV   4
#define FEAT    270          // TX*RXS
#define FEATP   288          // FEAT padded to mult of 32
#define NDBC    64           // 48 (DT_RANK+2*D_STATE) padded to mult of 64
#define NOUT    51           // N_PERSON*N_KP*D_COORD

typedef __attribute__((ext_vector_type(16))) _Float16 v16h;
typedef __attribute__((ext_vector_type(8)))  float    v8f;

// ---------------- input index map (setup_inputs dict order, depth-first) ----
enum {
  I_X = 0, I_PROJ_W, I_PROJ_B,
  I_UP_W, I_UP_B, I_DOWN_W, I_DOWN_B, I_LN_W, I_LN_B,
  I_STM0 = 9,          // 4 blocks x 10 tensors
  I_DS_W = 49, I_DS_B = 50,
  I_LTM0 = 51,         // 2 blocks x 10 tensors
  I_UPC_W = 71, I_UPC_B, I_FN_W, I_FN_B, I_FC1_W, I_FC1_B, I_FC2_W, I_FC2_B
};
// per-mamba offsets: 0 in_proj, 1 conv_w, 2 conv_b, 3 x_proj, 4 dt_w, 5 dt_b,
//                    6 A_log, 7 D, 8 out_proj, 9 norm_w

// ------- weight pack: f32 [K,N] -> f16 [Npad,Kpad] (transposed, zero padded)
__global__ void pack_wt_k(const float* __restrict__ W, _Float16* __restrict__ Wt,
                          int K, int N, int Kpad, int Npad) {
  int idx = blockIdx.x * blockDim.x + threadIdx.x;
  if (idx >= Npad * Kpad) return;
  int n = idx / Kpad, k = idx % Kpad;
  Wt[idx] = (k < K && n < N) ? (_Float16)W[(size_t)k * N + n] : (_Float16)0.0f;
}

// ------- gather: x[B,3,90,L] -> XF[B*L, 288] (cols 270..287 zero) -------
__global__ void gather_xf_k(const float* __restrict__ x, float* __restrict__ XF) {
  int idx = blockIdx.x * blockDim.x + threadIdx.x;
  if (idx >= BATCH * SEQ_L * FEATP) return;
  int f = idx % FEATP;
  int l = (idx / FEATP) % SEQ_L;
  int b = idx / (FEATP * SEQ_L);
  XF[idx] = (f < FEAT)
      ? x[(((size_t)b * 3 + f / 90) * 90 + f % 90) * SEQ_L + l]
      : 0.0f;
}

// ---------------- WMMA GEMM ---------------------------------------------
// C[M,N] = act(A[M,lda](cols 0..Klim) @ Wt^T + bias) + res
// Wt: f16 [N, Kpad], N % 64 == 0, M % 16 == 0, Kloop % 32 == 0, Klim % 16 == 0.
// Each wave computes a 16(M) x 64(N) strip: one A fragment feeds 4 WMMAs.
// act: 0 none, 1 silu, 2 softplus.
template <bool FULLK>
__device__ __forceinline__ void gemm_mainloop(const float* __restrict__ arow,
                                              const _Float16* __restrict__ wp,
                                              size_t wstep, int Kloop, int Klim, int lh,
                                              v8f& acc0, v8f& acc1, v8f& acc2, v8f& acc3) {
  for (int k0 = 0; k0 < Kloop; k0 += 32) {
    int kb = k0 + lh * 16;
    v16h af = {};
    if (FULLK || kb < Klim) {               // FULLK: no guard, straight-line loads
      const float4* ap = (const float4*)(arow + kb);
      float4 q0 = ap[0], q1 = ap[1], q2 = ap[2], q3 = ap[3];
      af[0]  = (_Float16)q0.x; af[1]  = (_Float16)q0.y;
      af[2]  = (_Float16)q0.z; af[3]  = (_Float16)q0.w;
      af[4]  = (_Float16)q1.x; af[5]  = (_Float16)q1.y;
      af[6]  = (_Float16)q1.z; af[7]  = (_Float16)q1.w;
      af[8]  = (_Float16)q2.x; af[9]  = (_Float16)q2.y;
      af[10] = (_Float16)q2.z; af[11] = (_Float16)q2.w;
      af[12] = (_Float16)q3.x; af[13] = (_Float16)q3.y;
      af[14] = (_Float16)q3.z; af[15] = (_Float16)q3.w;
    }
    if (FULLK && k0 + 32 < Kloop)
      __builtin_prefetch(arow + kb + 32, 0, 1);   // global_prefetch_b8 next A chunk
    v16h b0 = *(const v16h*)(wp + kb);
    v16h b1 = *(const v16h*)(wp + wstep + kb);
    v16h b2 = *(const v16h*)(wp + 2 * wstep + kb);
    v16h b3 = *(const v16h*)(wp + 3 * wstep + kb);
    acc0 = __builtin_amdgcn_wmma_f32_16x16x32_f16(false, af, false, b0, (short)0, acc0, false, false);
    acc1 = __builtin_amdgcn_wmma_f32_16x16x32_f16(false, af, false, b1, (short)0, acc1, false, false);
    acc2 = __builtin_amdgcn_wmma_f32_16x16x32_f16(false, af, false, b2, (short)0, acc2, false, false);
    acc3 = __builtin_amdgcn_wmma_f32_16x16x32_f16(false, af, false, b3, (short)0, acc3, false, false);
  }
}

__global__ void gemm_wmma_k(const float* __restrict__ A, int lda, int Kloop, int Klim,
                            const _Float16* __restrict__ Wt, int Kpad,
                            const float* __restrict__ bias,
                            const float* __restrict__ res,
                            float* __restrict__ C, int M, int N, int act) {
  int Mt = M >> 4, Ng = N >> 6;
  int w = blockIdx.x * (blockDim.x >> 5) + (threadIdx.x >> 5);
  if (w >= Mt * Ng) return;
  int ng = w % Ng;
  int m_tile = w / Ng;
  int lane = threadIdx.x & 31;
  int lr = lane & 15;       // row within A-frag / col within B-frag
  int lh = lane >> 4;       // which 16-wide K chunk

  const float*    arow = A  + (size_t)(m_tile * 16 + lr) * lda;
  const _Float16* wp   = Wt + (size_t)(ng * 64 + lr) * Kpad;
  const size_t wstep = (size_t)16 * Kpad;

  v8f acc0 = {}, acc1 = {}, acc2 = {}, acc3 = {};
  if (Klim >= Kloop)   // uniform branch: all hot GEMMs take the unguarded path
    gemm_mainloop<true >(arow, wp, wstep, Kloop, Klim, lh, acc0, acc1, acc2, acc3);
  else                 // only dt_proj (K=16 of Kloop=32) takes the guarded path
    gemm_mainloop<false>(arow, wp, wstep, Kloop, Klim, lh, acc0, acc1, acc2, acc3);

  v8f accs[4] = {acc0, acc1, acc2, acc3};
#pragma unroll
  for (int i = 0; i < 4; ++i) {
    int ncol = ng * 64 + i * 16 + lr;
    float bv = bias ? bias[ncol] : 0.0f;
#pragma unroll
    for (int r = 0; r < 8; ++r) {
      int orow = m_tile * 16 + lh * 8 + r;
      float v = accs[i][r] + bv;
      if (act == 1)      v = v / (1.0f + __expf(-v));                 // silu
      else if (act == 2) v = (v > 20.0f) ? v : log1pf(__expf(v));     // softplus
      if (res) v += res[(size_t)orow * N + ncol];
      C[(size_t)orow * N + ncol] = v;
    }
  }
}

// ---------------- elementwise: GSFM gate  G = v * sigmoid(gt) -----------
__global__ void gate_k(const float* __restrict__ UP, float* __restrict__ G, int M) {
  int idx = blockIdx.x * blockDim.x + threadIdx.x;
  if (idx >= M * DI) return;
  int row = idx / DI, c = idx % DI;
  float v = UP[(size_t)row * (2 * DI) + c];
  float g = UP[(size_t)row * (2 * DI) + DI + c];
  G[idx] = v * (1.0f / (1.0f + __expf(-g)));
}

// ---------------- causal depthwise conv1d + SiLU ------------------------
__global__ void conv_silu_k(const float* __restrict__ XZ, const float* __restrict__ cw,
                            const float* __restrict__ cb, float* __restrict__ U,
                            int B, int L) {
  int idx = blockIdx.x * blockDim.x + threadIdx.x;
  if (idx >= B * L * DI) return;
  int d = idx % DI;
  int row = idx / DI;
  int t = row % L, b = row / L;
  float acc = cb[d];
#pragma unroll
  for (int k = 0; k < DCONV; ++k) {
    int tt = t + k - (DCONV - 1);
    if (tt >= 0)
      acc += XZ[((size_t)(b * L + tt)) * (2 * DI) + d] * cw[d * DCONV + k];
  }
  U[idx] = acc / (1.0f + __expf(-acc));
}

// ---------------- selective scan (sequential over t), gated output ------
__global__ void ssm_scan_k(const float* __restrict__ DT, const float* __restrict__ U,
                           const float* __restrict__ DBC, const float* __restrict__ XZ,
                           const float* __restrict__ A_log, const float* __restrict__ Dp,
                           float* __restrict__ Y, int L) {
  int b = blockIdx.x, d = threadIdx.x;
  float h[DSN], Ad[DSN];
#pragma unroll
  for (int n = 0; n < DSN; ++n) { h[n] = 0.0f; Ad[n] = -__expf(A_log[d * DSN + n]); }
  float Dd = Dp[d];
  __shared__ float sB[DSN], sC[DSN];
  for (int t = 0; t < L; ++t) {
    size_t r = (size_t)(b * L + t);
    __syncthreads();
    if (d < DSN)           sB[d] = DBC[r * NDBC + DTR + d];
    else if (d < 2 * DSN)  sC[d - DSN] = DBC[r * NDBC + DTR + DSN + (d - DSN)];
    __syncthreads();
    float dt = DT[r * DI + d];
    float u  = U[r * DI + d];
    float y = 0.0f;
#pragma unroll
    for (int n = 0; n < DSN; ++n) {
      float dA = __expf(dt * Ad[n]);
      h[n] = dA * h[n] + dt * sB[n] * u;
      y += h[n] * sC[n];
    }
    y += u * Dd;
    float z = XZ[r * (2 * DI) + DI + d];
    y *= z / (1.0f + __expf(-z));
    Y[r * DI + d] = y;
  }
}

// ---------------- RMSNorm (one block per row of 256) --------------------
__global__ void rmsnorm_k(const float* __restrict__ X, const float* __restrict__ w,
                          float* __restrict__ O) {
  __shared__ float sred[DM];
  int row = blockIdx.x, tid = threadIdx.x;
  float v = X[(size_t)row * DM + tid];
  sred[tid] = v * v;
  __syncthreads();
  for (int s = DM >> 1; s > 0; s >>= 1) {
    if (tid < s) sred[tid] += sred[tid + s];
    __syncthreads();
  }
  float scale = rsqrtf(sred[0] / DM + 1e-6f);
  O[(size_t)row * DM + tid] = v * scale * w[tid];
}

// ---------------- LayerNorm (one block per row of 256) ------------------
__global__ void layernorm_k(const float* __restrict__ X, const float* __restrict__ w,
                            const float* __restrict__ bb, float* __restrict__ O) {
  __shared__ float sred[DM];
  int row = blockIdx.x, tid = threadIdx.x;
  float v = X[(size_t)row * DM + tid];
  sred[tid] = v;
  __syncthreads();
  for (int s = DM >> 1; s > 0; s >>= 1) {
    if (tid < s) sred[tid] += sred[tid + s];
    __syncthreads();
  }
  float m = sred[0] / DM;
  __syncthreads();
  float dlt = v - m;
  sred[tid] = dlt * dlt;
  __syncthreads();
  for (int s = DM >> 1; s > 0; s >>= 1) {
    if (tid < s) sred[tid] += sred[tid + s];
    __syncthreads();
  }
  float var = sred[0] / DM;
  O[(size_t)row * DM + tid] = dlt * rsqrtf(var + 1e-5f) * w[tid] + bb[tid];
}

// ---------------- downsample: einsum('btrd,dr->btd') + bias -------------
__global__ void downsample_k(const float* __restrict__ XS, const float* __restrict__ dw,
                             const float* __restrict__ db, float* __restrict__ XD) {
  int idx = blockIdx.x * blockDim.x + threadIdx.x;
  if (idx >= BATCH * SEQ_L2 * DM) return;
  int d = idx % DM;
  int row = idx / DM;
  int t = row % SEQ_L2, b = row / SEQ_L2;
  float acc = db[d];
#pragma unroll
  for (int r = 0; r < 4; ++r)
    acc += XS[((size_t)(b * SEQ_L + t * 4 + r)) * DM + d] * dw[d * 4 + r];
  XD[idx] = acc;
}

// ---------------- head: fuse last step, LN, fc1+GELU, fc2 ---------------
__global__ void head_k(const float* __restrict__ xs, const float* __restrict__ xl,
                       const float* __restrict__ xres,
                       const float* __restrict__ upw, const float* __restrict__ upb,
                       const float* __restrict__ fnw, const float* __restrict__ fnb,
                       const float* __restrict__ f1w, const float* __restrict__ f1b,
                       const float* __restrict__ f2w, const float* __restrict__ f2b,
                       float* __restrict__ out) {
  __shared__ float sred[DM], sv[DM], sh[DM / 2];
  int b = blockIdx.x, tid = threadIdx.x;
  size_t rl = (size_t)(b * SEQ_L + SEQ_L - 1) * DM;
  size_t r2 = (size_t)(b * SEQ_L2 + SEQ_L2 - 1) * DM;
  // fused[:, -1, :] = xs + xu + x_res ; xu(last) uses r = (L-1)%4 = 3
  float last = xs[rl + tid] + xres[rl + tid] +
               xl[r2 + tid] * upw[tid * 4 + 3] + upb[tid];
  sred[tid] = last;
  __syncthreads();
  for (int s = DM >> 1; s > 0; s >>= 1) { if (tid < s) sred[tid] += sred[tid + s]; __syncthreads(); }
  float m = sred[0] / DM;
  __syncthreads();
  float dlt = last - m;
  sred[tid] = dlt * dlt;
  __syncthreads();
  for (int s = DM >> 1; s > 0; s >>= 1) { if (tid < s) sred[tid] += sred[tid + s]; __syncthreads(); }
  float var = sred[0] / DM;
  sv[tid] = dlt * rsqrtf(var + 1e-5f) * fnw[tid] + fnb[tid];
  __syncthreads();
  if (tid < DM / 2) {
    float h = f1b[tid];
    for (int k = 0; k < DM; ++k) h += sv[k] * f1w[k * (DM / 2) + tid];
    sh[tid] = 0.5f * h * (1.0f + erff(h * 0.70710678f));   // exact GELU
  }
  __syncthreads();
  if (tid < NOUT) {
    float o = f2b[tid];
    for (int k = 0; k < DM / 2; ++k) o += sh[k] * f2w[k * NOUT + tid];
    out[b * NOUT + tid] = o;
  }
}

// ======================================================================
extern "C" void kernel_launch(void* const* d_in, const int* in_sizes, int n_in,
                              void* d_out, int out_size, void* d_ws, size_t ws_size,
                              hipStream_t stream) {
  (void)in_sizes; (void)n_in; (void)out_size; (void)ws_size;
  const int M1 = BATCH * SEQ_L;    // 8192
  const int M2 = BATCH * SEQ_L2;   // 2048

  auto F = [&](int i) { return (const float*)d_in[i]; };

  // ---- workspace carve-out (256B aligned slices) ----
  size_t off = 0;
  auto carve = [&](size_t bytes) {
    void* p = (char*)d_ws + off;
    off += (bytes + 255) & ~(size_t)255;
    return p;
  };
  // f16 packed weights (transposed, K and N zero-padded)
  _Float16* wt_proj = (_Float16*)carve((size_t)DM * FEATP * 2);
  _Float16* wt_up   = (_Float16*)carve((size_t)(4 * DM) * DM * 2);
  _Float16* wt_down = (_Float16*)carve((size_t)DM * DI * 2);
  _Float16 *wt_in[6], *wt_xp[6], *wt_dt[6], *wt_out[6];
  for (int j = 0; j < 6; ++j) {
    wt_in[j]  = (_Float16*)carve((size_t)(2 * DI) * DM * 2);
    wt_xp[j]  = (_Float16*)carve((size_t)NDBC * DI * 2);   // N 48->64
    wt_dt[j]  = (_Float16*)carve((size_t)DI * 32 * 2);     // K 16->32
    wt_out[j] = (_Float16*)carve((size_t)DM * DI * 2);
  }
  // f32 activations
  float* xf      = (float*)carve((size_t)M1 * FEATP * 4);
  float* e       = (float*)carve((size_t)M1 * DM * 4);
  float* buf1024 = (float*)carve((size_t)M1 * 2 * DI * 4);   // UP / XZ
  float* bufU    = (float*)carve((size_t)M1 * DI * 4);       // gate / u
  float* bufDT   = (float*)carve((size_t)M1 * DI * 4);
  float* bufY    = (float*)carve((size_t)M1 * DI * 4);
  float* dbc     = (float*)carve((size_t)M1 * NDBC * 4);
  float* xn      = (float*)carve((size_t)M1 * DM * 4);
  float* preln   = (float*)carve((size_t)M1 * DM * 4);
  float* xres    = (float*)carve((size_t)M1 * DM * 4);
  float* xsA     = (float*)carve((size_t)M1 * DM * 4);
  float* xsB     = (float*)carve((size_t)M1 * DM * 4);
  float* xdA     = (float*)carve((size_t)M2 * DM * 4);
  float* xdB     = (float*)carve((size_t)M2 * DM * 4);

  auto pack = [&](const float* W, _Float16* Wt, int K, int N, int Kpad, int Npad) {
    int total = Npad * Kpad;
    pack_wt_k<<<(total + 255) / 256, 256, 0, stream>>>(W, Wt, K, N, Kpad, Npad);
  };
  // N here is the PADDED output width (multiple of 64)
  auto gemm = [&](const float* A, int lda, int Kloop, int Klim,
                  const _Float16* Wt, int Kpad, const float* bias,
                  const float* res, float* C, int M, int N, int act) {
    int tiles = (M / 16) * (N / 64);
    gemm_wmma_k<<<(tiles + 7) / 8, 256, 0, stream>>>(A, lda, Kloop, Klim, Wt, Kpad,
                                                     bias, res, C, M, N, act);
  };

  // ---- pack all GEMM weights ----
  pack(F(I_PROJ_W), wt_proj, FEAT, DM, FEATP, DM);
  pack(F(I_UP_W),   wt_up,   DM, 4 * DM, DM, 4 * DM);
  pack(F(I_DOWN_W), wt_down, DI, DM, DI, DM);
  for (int j = 0; j < 6; ++j) {
    int base = (j < 4) ? (I_STM0 + 10 * j) : (I_LTM0 + 10 * (j - 4));
    pack(F(base + 0), wt_in[j],  DM, 2 * DI, DM, 2 * DI);  // in_proj  [256,1024]
    pack(F(base + 3), wt_xp[j],  DI, 48, DI, NDBC);        // x_proj   [512,48->64]
    pack(F(base + 4), wt_dt[j],  DTR, DI, 32, DI);         // dt_w     [16->32,512]
    pack(F(base + 8), wt_out[j], DI, DM, DI, DM);          // out_proj [512,256]
  }

  // ---- front end ----
  gather_xf_k<<<(M1 * FEATP + 255) / 256, 256, 0, stream>>>(F(I_X), xf);
  gemm(xf, FEATP, FEATP, FEATP, wt_proj, FEATP, F(I_PROJ_B), nullptr, e, M1, DM, 0);
  // GSFM
  gemm(e, DM, DM, DM, wt_up, DM, F(I_UP_B), nullptr, buf1024, M1, 4 * DM, 0);
  gate_k<<<(M1 * DI + 255) / 256, 256, 0, stream>>>(buf1024, bufU, M1);
  gemm(bufU, DI, DI, DI, wt_down, DI, F(I_DOWN_B), e, preln, M1, DM, 0);
  layernorm_k<<<M1, DM, 0, stream>>>(preln, F(I_LN_W), F(I_LN_B), xres);

  // ---- mamba block runner ----
  auto run_mamba = [&](const float* Xin, float* Xout, int j, int Mloc, int Lloc) {
    int base = (j < 4) ? (I_STM0 + 10 * j) : (I_LTM0 + 10 * (j - 4));
    rmsnorm_k<<<Mloc, DM, 0, stream>>>(Xin, F(base + 9), xn);
    gemm(xn, DM, DM, DM, wt_in[j], DM, nullptr, nullptr, buf1024, Mloc, 2 * DI, 0);
    conv_silu_k<<<(Mloc * DI + 255) / 256, 256, 0, stream>>>(
        buf1024, F(base + 1), F(base + 2), bufU, BATCH, Lloc);
    gemm(bufU, DI, DI, DI, wt_xp[j], DI, nullptr, nullptr, dbc, Mloc, NDBC, 0);
    gemm(dbc, NDBC, 32, DTR, wt_dt[j], 32, F(base + 5), nullptr, bufDT, Mloc, DI, 2);
    ssm_scan_k<<<BATCH, DI, 0, stream>>>(bufDT, bufU, dbc, buf1024,
                                         F(base + 6), F(base + 7), bufY, Lloc);
    gemm(bufY, DI, DI, DI, wt_out[j], DI, nullptr, Xin, Xout, Mloc, DM, 0);
  };

  // STM chain (L=2048): xres -> A -> B -> A -> B
  run_mamba(xres, xsA, 0, M1, SEQ_L);
  run_mamba(xsA,  xsB, 1, M1, SEQ_L);
  run_mamba(xsB,  xsA, 2, M1, SEQ_L);
  run_mamba(xsA,  xsB, 3, M1, SEQ_L);   // xs = xsB

  // downsample -> LTM chain (L=512): xdA -> xdB -> xdA
  downsample_k<<<(M2 * DM + 255) / 256, 256, 0, stream>>>(xsB, F(I_DS_W), F(I_DS_B), xdA);
  run_mamba(xdA, xdB, 4, M2, SEQ_L2);
  run_mamba(xdB, xdA, 5, M2, SEQ_L2);   // xl = xdA

  // head (only last time step of fused sequence matters)
  head_k<<<BATCH, DM, 0, stream>>>(xsB, xdA, xres, F(I_UPC_W), F(I_UPC_B),
                                   F(I_FN_W), F(I_FN_B), F(I_FC1_W), F(I_FC1_B),
                                   F(I_FC2_W), F(I_FC2_B), (float*)d_out);
}
